// MGNOLayer_15951508537752
// MI455X (gfx1250) — matrix operand order repaired and checked by reading, not compiled, not run on hardware
//
#include <hip/hip_runtime.h>
#include <hip/hip_bf16.h>

// MGNO layer for MI455X (gfx1250, wave32, WMMA 16x16x32 f16, f32 accum).
// B=2, N=4096, C=64, ORDER=4, COORD=2, K=8. Tokens T = 8192.
//
// All WMMA operand fragments are kept in "fragment-ready" LDS layout:
// one contiguous 32B chunk (16 halves) per (k-tile, n-tile, lane), so a
// fragment load is 2x ds_load_b128 instead of 16x ds_load_u16.

typedef __attribute__((ext_vector_type(16))) _Float16 v16h;
typedef __attribute__((ext_vector_type(8)))  float    v8f;

#define TOK      8192
#define NPTS     4096
#define CC       64
#define KNN      8
#define EPSF     1e-8f
#define LN_EPSF  1e-6f

__device__ __forceinline__ float cleanf(float v) {
  if (v != v) return 0.0f;               // NaN -> 0
  return fminf(fmaxf(v, -1.0e6f), 1.0e6f);
}

__device__ __forceinline__ float gelu_tanh(float v) {
  const float k0 = 0.7978845608028654f;  // sqrt(2/pi)
  float v3 = v * v * v;
  return 0.5f * v * (1.0f + tanhf(k0 * (v + 0.044715f * v3)));
}

#define WMMA_F16(a, b, c) \
  __builtin_amdgcn_wmma_f32_16x16x32_f16(false, (a), false, (b), (short)0, (c), false, false)

// ---- fragment-ready LDS helpers (wave32 layouts per CDNA5 ISA 7.12.2) -----
// B fragment semantics: lanes 0-15 hold K=k0..k0+7,k0+16..23, lanes 16-31 K=k0+8..15,k0+24..31,
// half h -> K = k0 + 8*(lane/16) + h + (h&8), column N = n0 + lane%16.  Chunk index for frag
// (kt, nt, lane) = (kt*NT + nt)*32 + lane; 16 halves contiguous.
__device__ __forceinline__ v16h fragLd(const _Float16* base, int NT, int kt, int nt) {
  int lane = threadIdx.x & 31;
  return *(const v16h*)(base + (((kt * NT + nt) * 32 + lane) << 4));
}

// Fill frag-ready buffer from row-major f32 W[K][N] (N multiple of 16, K of 32).
template <int KT, int NT>
__device__ __forceinline__ void fillW(_Float16* dst, const float* __restrict__ W,
                                      int N, int nthreads) {
  for (int c = threadIdx.x; c < KT * NT * 32; c += nthreads) {
    int lane = c & 31;
    int tile = c >> 5;
    int nt = tile & (NT - 1);
    int kt = tile / NT;
    int g = lane >> 4, lc = lane & 15;
    const float* src = W + (kt * 32 + g * 8) * N + nt * 16 + lc;
    v16h v;
#pragma unroll
    for (int h = 0; h < 16; ++h) v[h] = (_Float16)src[(h + (h & 8)) * N];
    *(v16h*)(dst + (c << 4)) = v;
  }
}

// A fragment built from two contiguous 8-float groups (K=kb..kb+7, kb+16..kb+23).
__device__ __forceinline__ v16h buildA(const float* lo, const float* hi) {
  v16h a;
#pragma unroll
  for (int h = 0; h < 8; ++h) a[h] = (_Float16)lo[h];
#pragma unroll
  for (int h = 0; h < 8; ++h) a[8 + h] = (_Float16)hi[h];
  return a;
}
__device__ __forceinline__ v16h buildA_scaled(const float* lo, const float* hi, float s) {
  v16h a;
#pragma unroll
  for (int h = 0; h < 8; ++h) a[h] = (_Float16)(cleanf(lo[h]) * s);
#pragma unroll
  for (int h = 0; h < 8; ++h) a[8 + h] = (_Float16)(cleanf(hi[h]) * s);
  return a;
}

// Store one D-layout accumulator element (row, col) into an A-frag-ready f16
// buffer (so the next GEMM reads it back as contiguous chunks).
__device__ __forceinline__ void stA_frag(_Float16* A, int col, int row, float v) {
  int ks = col >> 5, q = col & 31;
  int g = (q >> 3) & 1;
  int h = (q & 7) + ((q >> 4) << 3);
  int lp = g * 16 + row;
  A[((ks * 32 + lp) << 4) + h] = (_Float16)v;
}

// D store to row-major f32 LDS tile: VGPR r -> row r / r+8, N = lane%16.
__device__ __forceinline__ void stD(float* D, int ldn, int n0, v8f d) {
  int lane = threadIdx.x & 31;
  int n = n0 + (lane & 15);
  int mo = (lane >> 4) ? 8 : 0;
#pragma unroll
  for (int r = 0; r < 8; ++r) D[(mo + r) * ldn + n] = d[r];
}

// ===========================================================================
// Kernel 1: multipole.  A'(8192x320) @ W'(320x64) with per-token scalars c_o,
// then LayerNorm.  WG=128 (4 waves), wave = 16 tokens, grid = 128.
// ===========================================================================
__global__ __launch_bounds__(128) void k_multipole(
    const float* __restrict__ x, const float* __restrict__ pos,
    const float* __restrict__ mw, const float* __restrict__ oscale,
    const float* __restrict__ lns, const float* __restrict__ lnb,
    float* __restrict__ mp_out)
{
  __shared__ __align__(32) _Float16 Wh[10 * 4 * 32 * 16]; // 40 KB frag-ready W'
  __shared__ float Dt[4][16 * 64];                        // 16 KB per-wave tiles

  int tid = threadIdx.x, wave = tid >> 5, lane = tid & 31;

  // W'[64*o + i][j] = mw[(i*64 + j)*5 + o], stored frag-ready (KT=10, NT=4).
  for (int c = tid; c < 10 * 4 * 32; c += 128) {
    int fl = c & 31;
    int tile = c >> 5;
    int nt = tile & 3, kt = tile >> 2;
    int g = fl >> 4, lc = fl & 15;
    int col = nt * 16 + lc;
    v16h v;
#pragma unroll
    for (int h = 0; h < 16; ++h) {
      int krow = kt * 32 + g * 8 + h + (h & 8);
      int o = krow >> 6, i = krow & 63;
      v[h] = (_Float16)mw[(i * 64 + col) * 5 + o];
    }
    *(v16h*)(Wh + (c << 4)) = v;
  }
  __syncthreads();

  int t0 = (blockIdx.x * 4 + wave) * 16;
  int m  = lane & 15;
  int t  = t0 + m;
  const float*  xr = x + t * CC;
  const float4* x4 = (const float4*)xr;

  // per-row multipole scalars (both lane groups compute for row m)
  float p0 = cleanf(pos[t * 2 + 0]);
  float p1 = cleanf(pos[t * 2 + 1]);
  float r  = fmaxf(sqrtf(p0 * p0 + p1 * p1), EPSF);
  float th = atan2f(p1, p0 + EPSF);
  float xn2 = 0.0f;
  for (int j4 = 0; j4 < 16; ++j4) {
    float4 u = x4[j4];
    float a0 = cleanf(u.x), a1 = cleanf(u.y), a2 = cleanf(u.z), a3 = cleanf(u.w);
    xn2 += a0 * a0 + a1 * a1 + a2 * a2 + a3 * a3;
  }
  float xnrm = sqrtf(xn2);

  float c[5];
  float ro = 1.0f;
#pragma unroll
  for (int o = 0; o < 5; ++o) {
    float radial = tanhf(ro / (1.0f + ro));
    float ang    = cosf((float)o * th);
    float ra     = radial * ang;
    float s      = fminf(fmaxf(oscale[o], 0.01f), 1.0f);
    c[o] = ra / (fabsf(ra) * xnrm + EPSF) * s / (float)(o + 1);
    ro *= r;
  }

  int kb = (lane >> 4) ? 8 : 0;
  v8f acc[4];
#pragma unroll
  for (int nt = 0; nt < 4; ++nt)
#pragma unroll
    for (int q = 0; q < 8; ++q) acc[nt][q] = 0.0f;

#pragma unroll
  for (int kt = 0; kt < 10; ++kt) {
    int o = kt >> 1;
    int base = (kt & 1) * 32 + kb;          // in {0,8,32,40}
    float lo[8], hi[8];
    *(float4*)(lo)     = x4[(base >> 2)];
    *(float4*)(lo + 4) = x4[(base >> 2) + 1];
    *(float4*)(hi)     = x4[(base >> 2) + 4];
    *(float4*)(hi + 4) = x4[(base >> 2) + 5];
    v16h a = buildA_scaled(lo, hi, c[o]);
#pragma unroll
    for (int nt = 0; nt < 4; ++nt)
      acc[nt] = WMMA_F16(a, fragLd(Wh, 4, kt, nt), acc[nt]);
  }

  float* D = Dt[wave];
#pragma unroll
  for (int nt = 0; nt < 4; ++nt) stD(D, 64, nt * 16, acc[nt]);
  __syncthreads();

  if (lane < 16) {
    const float4* D4 = (const float4*)(D + m * 64);
    float mu = 0.0f;
    for (int j4 = 0; j4 < 16; ++j4) {
      float4 u = D4[j4];
      mu += u.x + u.y + u.z + u.w;
    }
    mu *= (1.0f / 64.0f);
    float var = 0.0f;
    for (int j4 = 0; j4 < 16; ++j4) {
      float4 u = D4[j4];
      float d0 = u.x - mu, d1 = u.y - mu, d2 = u.z - mu, d3 = u.w - mu;
      var += d0 * d0 + d1 * d1 + d2 * d2 + d3 * d3;
    }
    var *= (1.0f / 64.0f);
    float inv = rsqrtf(var + LN_EPSF);
    float4* out4 = (float4*)(mp_out + t * CC);
    for (int j4 = 0; j4 < 16; ++j4) {
      float4 u = D4[j4];
      float4 xv = x4[j4];
      float4 s = *(const float4*)(lns + j4 * 4);
      float4 bb = *(const float4*)(lnb + j4 * 4);
      float4 y;
      y.x = (u.x - mu) * inv * s.x + bb.x;
      y.y = (u.y - mu) * inv * s.y + bb.y;
      y.z = (u.z - mu) * inv * s.z + bb.z;
      y.w = (u.w - mu) * inv * s.w + bb.w;
      if (y.x != y.x) y.x = cleanf(xv.x);
      if (y.y != y.y) y.y = cleanf(xv.y);
      if (y.z != y.z) y.z = cleanf(xv.z);
      if (y.w != y.w) y.w = cleanf(xv.w);
      out4[j4] = y;
    }
  }
}

// ===========================================================================
// Kernel 2: 8-NN per query (self excluded).  One wave per query; per-lane
// sorted top-8 over 4096/32 candidates, then 5-step shuffle merge (wave32).
// WG=256 (8 waves), grid=1024.
// ===========================================================================
__global__ __launch_bounds__(256) void k_topk(
    const float* __restrict__ pos, int* __restrict__ nbr)
{
  int wave = threadIdx.x >> 5, lane = threadIdx.x & 31;
  int q = blockIdx.x * 8 + wave;
  int b = q >> 12, n = q & (NPTS - 1);
  const float* pb = pos + b * NPTS * 2;
  float qx = pb[n * 2], qy = pb[n * 2 + 1];

  float bd[KNN]; int bi[KNN];
#pragma unroll
  for (int k = 0; k < KNN; ++k) { bd[k] = 3.4e38f; bi[k] = 0; }

  for (int j = lane; j < NPTS; j += 32) {
    if (j == n) continue;
    float dx = pb[j * 2] - qx, dy = pb[j * 2 + 1] - qy;
    float d = dx * dx + dy * dy;
    if (d < bd[KNN - 1]) {
      int k = KNN - 1;
      while (k > 0 && bd[k - 1] > d) { bd[k] = bd[k - 1]; bi[k] = bi[k - 1]; --k; }
      bd[k] = d; bi[k] = j;
    }
  }

#pragma unroll
  for (int off = 16; off >= 1; off >>= 1) {
    float od[KNN]; int oi[KNN];
#pragma unroll
    for (int k = 0; k < KNN; ++k) {
      od[k] = __shfl_down(bd[k], off, 32);
      oi[k] = __shfl_down(bi[k], off, 32);
    }
    float md[KNN]; int mi[KNN];
    int ia = 0, ib = 0;
#pragma unroll
    for (int k = 0; k < KNN; ++k) {
      bool ta = (ib >= KNN) || (ia < KNN && bd[ia] <= od[ib]);
      if (ta) { md[k] = bd[ia]; mi[k] = bi[ia]; ++ia; }
      else    { md[k] = od[ib]; mi[k] = oi[ib]; ++ib; }
    }
#pragma unroll
    for (int k = 0; k < KNN; ++k) { bd[k] = md[k]; bi[k] = mi[k]; }
  }

  if (lane == 0) {
#pragma unroll
    for (int k = 0; k < KNN; ++k) nbr[q * KNN + k] = bi[k];
  }
}

// ===========================================================================
// Kernel 3: messages.  65536 rows of comb[128]=[x_self|x_nbr];
// GEMM1 (K=128,N=64) -> gelu -> GEMM2 (K=64,N=64) -> mean over 8 nbrs + b2.
// WG=128 (4 waves), wave = 16 rows = 2 queries, grid = 1024.
// ===========================================================================
__global__ __launch_bounds__(128) void k_messages(
    const float* __restrict__ x, const int* __restrict__ nbr,
    const float* __restrict__ W1, const float* __restrict__ b1,
    const float* __restrict__ W2, const float* __restrict__ b2,
    float* __restrict__ lm)
{
  __shared__ __align__(32) _Float16 W1h[4 * 4 * 32 * 16]; // 16 KB frag-ready
  __shared__ __align__(32) _Float16 W2h[2 * 4 * 32 * 16]; //  8 KB frag-ready
  __shared__ __align__(32) _Float16 Af[4][2 * 32 * 16];   //  8 KB A-frag hidden
  __shared__ float Dt[4][16 * 64];                        // 16 KB f32 tiles

  int tid = threadIdx.x, wave = tid >> 5, lane = tid & 31;
  fillW<4, 4>(W1h, W1, 64, 128);
  fillW<2, 4>(W2h, W2, 64, 128);
  __syncthreads();

  int row0 = (blockIdx.x * 4 + wave) * 16;  // rows flattened as t*8 + k
  int m    = lane & 15;
  int row  = row0 + m;
  int t    = row >> 3;          // global token
  int b    = t >> 12;
  int nb   = nbr[row];          // neighbor index within batch
  const float* xs = x + t * CC;
  const float* xn = x + (b * NPTS + nb) * CC;
  __builtin_prefetch(xn, 0, 0);             // global_prefetch_b8 (gathered row)
  int kb = (lane >> 4) ? 8 : 0;

  v8f acc[4];
#pragma unroll
  for (int nt = 0; nt < 4; ++nt)
#pragma unroll
    for (int q = 0; q < 8; ++q) acc[nt][q] = 0.0f;

#pragma unroll
  for (int ks = 0; ks < 4; ++ks) {
    int g0 = ks * 32 + kb, g1 = g0 + 16;    // 8-float groups, never split at 64
    const float* s0 = (g0 < 64) ? xs + g0 : xn + g0 - 64;
    const float* s1 = (g1 < 64) ? xs + g1 : xn + g1 - 64;
    float lo[8], hi[8];
    *(float4*)(lo)     = *(const float4*)(s0);
    *(float4*)(lo + 4) = *(const float4*)(s0 + 4);
    *(float4*)(hi)     = *(const float4*)(s1);
    *(float4*)(hi + 4) = *(const float4*)(s1 + 4);
    v16h a = buildA(lo, hi);
#pragma unroll
    for (int nt = 0; nt < 4; ++nt)
      acc[nt] = WMMA_F16(a, fragLd(W1h, 4, ks, nt), acc[nt]);
  }

  // b1 + gelu -> A-frag-ready hidden tile
  _Float16* A = Af[wave];
  {
    int n16 = lane & 15, mo = (lane >> 4) ? 8 : 0;
#pragma unroll
    for (int nt = 0; nt < 4; ++nt)
#pragma unroll
      for (int r = 0; r < 8; ++r) {
        int col = nt * 16 + n16;
        stA_frag(A, col, mo + r, gelu_tanh(acc[nt][r] + b1[col]));
      }
  }
  __syncthreads();

  v8f acc2[4];
#pragma unroll
  for (int nt = 0; nt < 4; ++nt)
#pragma unroll
    for (int q = 0; q < 8; ++q) acc2[nt][q] = 0.0f;
#pragma unroll
  for (int ks = 0; ks < 2; ++ks) {
    v16h a2 = *(const v16h*)(A + ((ks * 32 + lane) << 4));
#pragma unroll
    for (int nt = 0; nt < 4; ++nt)
      acc2[nt] = WMMA_F16(a2, fragLd(W2h, 4, ks, nt), acc2[nt]);
  }

  float* D = Dt[wave];
#pragma unroll
  for (int nt = 0; nt < 4; ++nt) stD(D, 64, nt * 16, acc2[nt]);
  __syncthreads();

  // mean over 8 neighbor rows per query; lane reduces a contiguous 4-col slice
  {
    int qi = lane >> 4;               // 0 or 1
    int j0 = (lane & 15) * 4;
    int q = (row0 >> 3) + qi;         // global token (row0 multiple of 16)
    float4 s = {0.0f, 0.0f, 0.0f, 0.0f};
#pragma unroll
    for (int k = 0; k < 8; ++k) {
      float4 u = *(const float4*)(D + (qi * 8 + k) * 64 + j0);
      s.x += u.x; s.y += u.y; s.z += u.z; s.w += u.w;
    }
    float4 bb = *(const float4*)(b2 + j0);
    float4 o;
    o.x = s.x * 0.125f + bb.x;
    o.y = s.y * 0.125f + bb.y;
    o.z = s.z * 0.125f + bb.z;
    o.w = s.w * 0.125f + bb.w;
    *(float4*)(lm + q * CC + j0) = o;
  }
}

// ===========================================================================
// Kernel 4: update MLP + residual + LayerNorm.
// u=[mp|lm] (8192x128) @ W1(128x128) -> gelu -> @ W2(128x64) -> +x -> LN.
// WG=64 (2 waves), wave = 16 tokens, grid = 256.
// ===========================================================================
__global__ __launch_bounds__(64) void k_update(
    const float* __restrict__ x,
    const float* __restrict__ mp, const float* __restrict__ lm,
    const float* __restrict__ W1, const float* __restrict__ b1,
    const float* __restrict__ W2, const float* __restrict__ b2,
    const float* __restrict__ lns, const float* __restrict__ lnb,
    float* __restrict__ out)
{
  __shared__ __align__(32) _Float16 W1h[4 * 8 * 32 * 16]; // 32 KB frag-ready
  __shared__ __align__(32) _Float16 W2h[4 * 4 * 32 * 16]; // 16 KB frag-ready
  __shared__ __align__(32) unsigned char HB[2][4096];     //  8 KB H-frag / O-tile

  int tid = threadIdx.x, wave = tid >> 5, lane = tid & 31;
  fillW<4, 8>(W1h, W1, 128, 64);
  fillW<4, 4>(W2h, W2, 64, 64);
  __syncthreads();

  int t0 = (blockIdx.x * 2 + wave) * 16;
  int m  = lane & 15;
  int t  = t0 + m;
  const float* us = mp + t * CC;
  const float* un = lm + t * CC;
  int kb = (lane >> 4) ? 8 : 0;

  v8f acc[8];
#pragma unroll
  for (int nt = 0; nt < 8; ++nt)
#pragma unroll
    for (int q = 0; q < 8; ++q) acc[nt][q] = 0.0f;

#pragma unroll
  for (int ks = 0; ks < 4; ++ks) {
    int g0 = ks * 32 + kb, g1 = g0 + 16;
    const float* s0 = (g0 < 64) ? us + g0 : un + g0 - 64;
    const float* s1 = (g1 < 64) ? us + g1 : un + g1 - 64;
    float lo[8], hi[8];
    *(float4*)(lo)     = *(const float4*)(s0);
    *(float4*)(lo + 4) = *(const float4*)(s0 + 4);
    *(float4*)(hi)     = *(const float4*)(s1);
    *(float4*)(hi + 4) = *(const float4*)(s1 + 4);
    v16h a = buildA(lo, hi);
#pragma unroll
    for (int nt = 0; nt < 8; ++nt)
      acc[nt] = WMMA_F16(a, fragLd(W1h, 8, ks, nt), acc[nt]);
  }

  _Float16* H = (_Float16*)HB[wave];   // A-frag-ready 16x128 f16 hidden tile
  {
    int n16 = lane & 15, mo = (lane >> 4) ? 8 : 0;
#pragma unroll
    for (int nt = 0; nt < 8; ++nt)
#pragma unroll
      for (int r = 0; r < 8; ++r) {
        int col = nt * 16 + n16;
        stA_frag(H, col, mo + r, gelu_tanh(acc[nt][r] + b1[col]));
      }
  }
  __syncthreads();

  // load all A-frags before reusing HB as the f32 output tile
  v16h a2[4];
#pragma unroll
  for (int ks = 0; ks < 4; ++ks)
    a2[ks] = *(const v16h*)(H + ((ks * 32 + lane) << 4));
  __syncthreads();

  v8f acc2[4];
#pragma unroll
  for (int nt = 0; nt < 4; ++nt)
#pragma unroll
    for (int q = 0; q < 8; ++q) acc2[nt][q] = 0.0f;
#pragma unroll
  for (int ks = 0; ks < 4; ++ks)
#pragma unroll
    for (int nt = 0; nt < 4; ++nt)
      acc2[nt] = WMMA_F16(a2[ks], fragLd(W2h, 4, ks, nt), acc2[nt]);

  float* O = (float*)HB[wave];         // 16 x 64 f32 output tile (reuse)
#pragma unroll
  for (int nt = 0; nt < 4; ++nt) stD(O, 64, nt * 16, acc2[nt]);
  __syncthreads();

  if (lane < 16) {
    const float4* xr4 = (const float4*)(x + t * CC);
    const float4* O4  = (const float4*)(O + m * 64);
    float vrow[64];
    float mu = 0.0f;
#pragma unroll
    for (int j4 = 0; j4 < 16; ++j4) {
      float4 xv = xr4[j4];
      float4 ov = O4[j4];
      float4 bb = *(const float4*)(b2 + j4 * 4);
      vrow[j4 * 4 + 0] = xv.x + ov.x + bb.x;
      vrow[j4 * 4 + 1] = xv.y + ov.y + bb.y;
      vrow[j4 * 4 + 2] = xv.z + ov.z + bb.z;
      vrow[j4 * 4 + 3] = xv.w + ov.w + bb.w;
      mu += vrow[j4 * 4 + 0] + vrow[j4 * 4 + 1] + vrow[j4 * 4 + 2] + vrow[j4 * 4 + 3];
    }
    mu *= (1.0f / 64.0f);
    float var = 0.0f;
#pragma unroll
    for (int j = 0; j < 64; ++j) { float dv = vrow[j] - mu; var += dv * dv; }
    var *= (1.0f / 64.0f);
    float inv = rsqrtf(var + LN_EPSF);
    float4* out4 = (float4*)(out + t * CC);
#pragma unroll
    for (int j4 = 0; j4 < 16; ++j4) {
      float4 s = *(const float4*)(lns + j4 * 4);
      float4 bb = *(const float4*)(lnb + j4 * 4);
      float4 y;
      y.x = (vrow[j4 * 4 + 0] - mu) * inv * s.x + bb.x;
      y.y = (vrow[j4 * 4 + 1] - mu) * inv * s.y + bb.y;
      y.z = (vrow[j4 * 4 + 2] - mu) * inv * s.z + bb.z;
      y.w = (vrow[j4 * 4 + 3] - mu) * inv * s.w + bb.w;
      out4[j4] = y;
    }
  }
}

// ===========================================================================
extern "C" void kernel_launch(void* const* d_in, const int* in_sizes, int n_in,
                              void* d_out, int out_size, void* d_ws, size_t ws_size,
                              hipStream_t stream) {
  const float* x      = (const float*)d_in[0];   // (B,N,C)
  const float* pos    = (const float*)d_in[1];   // (B,N,2)
  const float* mw     = (const float*)d_in[2];   // (C,C,5)
  const float* oscale = (const float*)d_in[3];   // (5)
  const float* mplns  = (const float*)d_in[4];
  const float* mplnb  = (const float*)d_in[5];
  const float* msgW1  = (const float*)d_in[6];   // (128,64)
  const float* msgb1  = (const float*)d_in[7];
  const float* msgW2  = (const float*)d_in[8];   // (64,64)
  const float* msgb2  = (const float*)d_in[9];
  const float* updW1  = (const float*)d_in[10];  // (128,128)
  const float* updb1  = (const float*)d_in[11];
  const float* updW2  = (const float*)d_in[12];  // (128,64)
  const float* updb2  = (const float*)d_in[13];
  const float* olns   = (const float*)d_in[14];
  const float* olnb   = (const float*)d_in[15];

  float* ws_mp  = (float*)d_ws;                      // 8192*64 f32
  float* ws_lm  = ws_mp + TOK * CC;                  // 8192*64 f32
  int*   ws_nbr = (int*)(ws_lm + TOK * CC);          // 8192*8  i32

  k_multipole<<<128, 128, 0, stream>>>(x, pos, mw, oscale, mplns, mplnb, ws_mp);
  k_topk<<<1024, 256, 0, stream>>>(pos, ws_nbr);
  k_messages<<<1024, 128, 0, stream>>>(x, ws_nbr, msgW1, msgb1, msgW2, msgb2, ws_lm);
  k_update<<<256, 64, 0, stream>>>(x, ws_mp, ws_lm, updW1, updb1, updW2, updb2,
                                   olns, olnb, (float*)d_out);
}
